// HybridLaplacianBlock_15255723835693
// MI455X (gfx1250) — matrix operand order repaired and checked by reading, not compiled, not run on hardware
//
#include <hip/hip_runtime.h>
#include <hip/hip_bf16.h>
#include <math.h>

typedef __attribute__((ext_vector_type(16))) _Float16 v16h;
typedef __attribute__((ext_vector_type(2)))  __fp16   fp16x2;
typedef __attribute__((ext_vector_type(8)))  float    v8f;

// Hardware tanh (gfx1250 V_TANH_F32, a TRANS32 op) — guard the builtin name.
#if __has_builtin(__builtin_amdgcn_tanhf)
  #define HW_TANH(x) __builtin_amdgcn_tanhf(x)
#elif __has_builtin(__builtin_amdgcn_tanh_f32)
  #define HW_TANH(x) __builtin_amdgcn_tanh_f32(x)
#else
  #define HW_TANH(x) tanhf(x)
#endif

// VALU butterfly add via DPP16 ROW_XMASK (0x160 + mask): XOR pattern within a
// 16-lane DPP row -> no LDS traffic, no dscnt waits, no EXEC branches.
#define DPP_XADD(x, ctrl)                                                     \
  ((x) + __builtin_bit_cast(float, __builtin_amdgcn_update_dpp(               \
             0, __builtin_bit_cast(int, (x)), (ctrl), 0xf, 0xf, true)))

#define NGRID   1048576      // N (grid points per batch row), power of two
#define NBATCH  4            // B
#define HDIM    32           // hidden width
#define NPOINTS (NGRID * NBATCH)
#define WAVES   8            // waves per block (256 threads, wave32)
#define TPW     8            // 16-point tiles per wave
#define NBLOCKS (NPOINTS / (WAVES * TPW * 16))   // 4096, exact cover

__launch_bounds__(256)
__global__ void hybrid_lap_mlp_wmma(const float* __restrict__ u,
                                    const float* __restrict__ W1,
                                    const float* __restrict__ B1,
                                    const float* __restrict__ W2,
                                    const float* __restrict__ B2,
                                    const float* __restrict__ W3,
                                    const float* __restrict__ B3,
                                    float* __restrict__ out) {
  __shared__ float sW1[HDIM], sB1[HDIM], sW2[HDIM * HDIM], sB2[HDIM], sW3[HDIM];
  __shared__ float sB3;

  const int tid = threadIdx.x;
  if (tid < HDIM) {
    sW1[tid] = W1[tid]; sB1[tid] = B1[tid];
    sB2[tid] = B2[tid]; sW3[tid] = W3[tid];
  }
  for (int i = tid; i < HDIM * HDIM; i += 256) sW2[i] = W2[i];
  if (tid == 0) sB3 = B3[0];
  __syncthreads();

  const int lane = tid & 31;
  const int wave = tid >> 5;
  const int half = lane >> 4;   // lane half selects K sub-range in A/B layouts
  const int lm   = lane & 15;   // row m (A) / column n (B, C, D)

  // ---- per-lane A-layout K map, W1/b1 slices hoisted into registers ----
  // 16-bit A 16x32: lanes 0-15 hold K {0..7,16..23}; lanes 16-31 K {8..15,24..31}.
  float w1r[16], b1r[16];
  #pragma unroll
  for (int j = 0; j < 16; ++j) {
    const int k = ((j < 8) ? (j & 14) : (16 + ((j - 8) & 14)))
                  + 8 * half + (j & 1);
    w1r[j] = sW1[k];
    b1r[j] = sB1[k];
  }

  // ---- B operands: W2 in f16, resident in VGPRs for the whole kernel ----
  // 16-bit B (32x16) layout: lane (half h, col n) holds halves j=0..15 of
  // elements (K = 8*h + j for j<8, 16 + 8*h + (j-8) for j>=8, N = n).  bOp0 -> W2 cols 0-15, bOp1 -> 16-31.
  union PackV { v16h v; fp16x2 p[8]; };
  PackV bOp0, bOp1;
  #pragma unroll
  for (int v = 0; v < 8; ++v) {
    const int k0 = ((v < 4) ? (2 * v) : (16 + 2 * (v - 4))) + 8 * half;
    bOp0.p[v] = __builtin_amdgcn_cvt_pkrtz(sW2[k0 * HDIM + lm],
                                           sW2[(k0 + 1) * HDIM + lm]);
    bOp1.p[v] = __builtin_amdgcn_cvt_pkrtz(sW2[k0 * HDIM + lm + 16],
                                           sW2[(k0 + 1) * HDIM + lm + 16]);
  }
  // C operand carries the layer-2 bias: f32 C/D layout has N = lane column,
  // all 8 row-VGPRs share the same column -> broadcast b2[N] down each reg.
  v8f cB0, cB1;
  #pragma unroll
  for (int r = 0; r < 8; ++r) { cB0[r] = sB2[lm]; cB1[r] = sB2[lm + 16]; }

  const float w3a = sW3[lm];        // W3 for k = lm      (from d0)
  const float w3b = sW3[lm + 16];   // W3 for k = lm + 16 (from d1)
  const float b3v = sB3;
  const int   sm  = lm + 8 * half;  // this lane's store row (valid when lm<8)

  for (int t = 0; t < TPW; ++t) {
    const long g = (long)blockIdx.x * (WAVES * TPW) + (long)t * WAVES + wave;
    const long base = g * 16;
    const long p    = base + lm;    // A-row load; both halves load row m = lm

    const float c = u[p];

    // ---- layer 1: fma + HW tanh straight into the WMMA A layout,
    //      packed pairwise with v_cvt_pk_rtz_f16_f32. ----
    PackV aOp;
    #pragma unroll
    for (int v = 0; v < 8; ++v) {
      const float h0 = HW_TANH(fmaf(c, w1r[2 * v],     b1r[2 * v]));
      const float h1 = HW_TANH(fmaf(c, w1r[2 * v + 1], b1r[2 * v + 1]));
      aOp.p[v] = __builtin_amdgcn_cvt_pkrtz(h0, h1);
    }

    // ---- layer 2: (16x32) @ (32x32) + b2 via two f16 WMMAs, f32 accum ----
    v8f d0 = __builtin_amdgcn_wmma_f32_16x16x32_f16(false, aOp.v, false, bOp0.v,
                                                    (short)0, cB0, false, false);
    v8f d1 = __builtin_amdgcn_wmma_f32_16x16x32_f16(false, aOp.v, false, bOp1.v,
                                                    (short)0, cB1, false, false);

    // ---- layer 3: HW tanh, scale by W3[N], DPP butterfly over 16 lanes ----
    // D layout: reg r holds rows (M = r) in lanes 0-15 and (M = r+8) in
    // lanes 16-31, column N = lm.  After the xmask 1/2/4/8 butterfly every
    // lane of a half holds row (r + 8*half)'s sum; lane lm==r keeps it.
    float myCorr = 0.0f;
    #pragma unroll
    for (int r = 0; r < 8; ++r) {
      float part = HW_TANH(d0[r]) * w3a + HW_TANH(d1[r]) * w3b;
      part = DPP_XADD(part, 0x161);   // xor 1
      part = DPP_XADD(part, 0x162);   // xor 2
      part = DPP_XADD(part, 0x164);   // xor 4
      part = DPP_XADD(part, 0x168);   // xor 8
      myCorr = (lm == r) ? part : myCorr;   // v_cndmask, no branch
    }

    // ---- stencil + store: lanes lm<8 own row sm = lm + 8*half ----
    // lanes 0-7 -> rows 0-7, lanes 16-23 -> rows 8-15: one coalesced store.
    if (lm < 8) {
      const long p2   = base + sm;
      const int  col  = (int)(p2 & (NGRID - 1));
      const float cc  = u[p2];
      const float lft = (col > 0)         ? u[p2 - 1] : 0.0f;
      const float rgt = (col < NGRID - 1) ? u[p2 + 1] : 0.0f;
      const float lap = lft - 2.0f * cc + rgt;   // [1,-2,1] stencil, dx = 1
      out[p2] = lap + myCorr + b3v;
    }
  }
}

extern "C" void kernel_launch(void* const* d_in, const int* in_sizes, int n_in,
                              void* d_out, int out_size, void* d_ws, size_t ws_size,
                              hipStream_t stream) {
  const float* u  = (const float*)d_in[0];
  const float* W1 = (const float*)d_in[1];
  const float* B1 = (const float*)d_in[2];
  const float* W2 = (const float*)d_in[3];
  const float* B2 = (const float*)d_in[4];
  const float* W3 = (const float*)d_in[5];
  const float* B3 = (const float*)d_in[6];
  float* out = (float*)d_out;
  (void)in_sizes; (void)n_in; (void)out_size; (void)d_ws; (void)ws_size;

  hipLaunchKernelGGL(hybrid_lap_mlp_wmma, dim3(NBLOCKS), dim3(256), 0, stream,
                     u, W1, B1, W2, B2, W3, B3, out);
}
